// Dlrm_86139864088679
// MI455X (gfx1250) — matrix operand order repaired and checked by reading, not compiled, not run on hardware
//
#include <hip/hip_runtime.h>

typedef __attribute__((ext_vector_type(16))) _Float16 v16h;
typedef __attribute__((ext_vector_type(8)))  _Float16 v8h;
typedef __attribute__((ext_vector_type(4)))  _Float16 v4h;
typedef __attribute__((ext_vector_type(8)))  float    v8f;

#define BATCH 16384
#define NTAB  26
#define VOCAB 50000
#define EDIM  128

__device__ __forceinline__ v16h cat16(v8h lo, v8h hi) {
  return __builtin_shufflevector(lo, hi, 0,1,2,3,4,5,6,7,8,9,10,11,12,13,14,15);
}

__device__ __forceinline__ v8f wmma_f16(v16h a, v16h b, v8f c) {
  return __builtin_amdgcn_wmma_f32_16x16x32_f16(false, a, false, b, (short)0, c, false, false);
}

// ---------------- weight convert + transpose: W(K,N) f32 -> Wt(N,Kpad) f16, zero-padded K ----
__global__ void Dlrm_wconv(const float* __restrict__ W, _Float16* __restrict__ Wt,
                           int K, int N, int Kpad) {
  size_t tid = (size_t)blockIdx.x * blockDim.x + threadIdx.x;
  if (tid >= (size_t)N * Kpad) return;
  int n  = (int)(tid / Kpad);
  int kp = (int)(tid % Kpad);
  float v = (kp < K) ? W[(size_t)kp * N + n] : 0.0f;
  Wt[tid] = (_Float16)v;
}

// ---------------- bottom layer 1: (B,13) @ (13,512) + bias, relu, f16 out -------------------
__global__ void Dlrm_bot1(const float* __restrict__ x, const float* __restrict__ W,
                          const float* __restrict__ bias, _Float16* __restrict__ out) {
  int tid = blockIdx.x * blockDim.x + threadIdx.x;   // BATCH*512 threads
  int b = tid >> 9;
  int n = tid & 511;
  float acc = bias[n];
#pragma unroll
  for (int k = 0; k < 13; ++k) acc += x[b * 13 + k] * W[k * 512 + n];
  out[(size_t)b * 512 + n] = (_Float16)(acc > 0.0f ? acc : 0.0f);
}

// ---------------- WMMA GEMM with async-to-LDS weight staging --------------------------------
// Block = 256 threads (8 waves). Block tile: M=128 (8 waves x 16 rows), N=64 (4 WMMA tiles).
// Weight 64x32 K-slices staged with global_load_async_to_lds_b128 into 4 LDS buffers,
// 2 copies in flight (s_wait_asynccnt 0x2). Execution-only barrier (raw signal/wait pair) so
// the A-fragment register double-buffer stays in flight across K-steps. sched_barrier batches
// the 8 ds_load_b128 ahead of the 4 WMMAs for partial dscnt waits.
#define BROW 40               // padded LDS row stride in halves (80 B: 16B-aligned, staggered)
#define BUFH (64 * BROW)      // halves per buffer (2560)
#define BUFB (BUFH * 2)       // bytes per buffer (5120)

__global__ __launch_bounds__(256)
void Dlrm_gemm(const _Float16* __restrict__ A, int lda,
               const _Float16* __restrict__ Wt, int ldb,
               const float* __restrict__ bias,
               _Float16* __restrict__ C, int ldc,
               int K) {
  __shared__ _Float16 Bs[4][64][BROW];   // 20480 B

  const int c    = threadIdx.x;          // 0..255
  const int wv   = c >> 5;
  const int lane = c & 31;
  const int hl   = lane & 15;
  const int hs   = lane >> 4;
  const int tq   = blockIdx.x;           // N strip (64 cols)
  const int tm   = blockIdx.y * 8 + wv;  // M tile (16 rows)

  // async staging: thread c copies the 16B chunk (row c>>2, chunk c&3) of the 64x32 tile
  const int brow_g = c >> 2;
  const int bchunk = c & 3;
  const uint32_t goff_base = (uint32_t)((brow_g * ldb + bchunk * 8) * 2);
  const uint32_t lds_base  = (uint32_t)(uintptr_t)(&Bs[0][0][0]) +
                             (uint32_t)(brow_g * (BROW * 2) + bchunk * 16);
  const void* gbase = (const void*)(Wt + (size_t)tq * 64 * ldb);

#define STAGE(ofs, kk)                                                      \
  asm volatile("global_load_async_to_lds_b128 %0, %1, %2"                   \
               :: "v"((uint32_t)(ofs)),                                     \
                  "v"((uint32_t)(goff_base + (uint32_t)((kk) * 2))),        \
                  "s"(gbase) : "memory");
// execution-only workgroup barrier: no VMEM-counter drain (async data is fenced explicitly)
#define BARRIER() asm volatile("s_barrier_signal -1\n\ts_barrier_wait -1" ::: "memory")

  const _Float16* arow = A + (size_t)(tm * 16 + hl) * lda + hs * 8;
  const _Float16* ldsf = &Bs[0][0][0] + hl * BROW + hs * 8;   // frag base inside a buffer

  const int S = K >> 5;                  // even (>= 8) for all layers
  STAGE(lds_base, 0)                     // buffer 0 <- k-step 0
  STAGE(lds_base + BUFB, 32)             // buffer 1 <- k-step 1

  v8h a0lo = *(const v8h*)(arow);        // A frag, even step
  v8h a0hi = *(const v8h*)(arow + 16);
  v8h a1lo = *(const v8h*)(arow + 32);   // A frag, odd step
  v8h a1hi = *(const v8h*)(arow + 48);

  v8f acc[4] = {v8f{}, v8f{}, v8f{}, v8f{}};

  auto kstep = [&](const _Float16* bbase, v16h a) {
    v16h bf[4];
#pragma unroll
    for (int sub = 0; sub < 4; ++sub) {
      const _Float16* bp = bbase + sub * 16 * BROW;
      bf[sub] = cat16(*(const v8h*)bp, *(const v8h*)(bp + 16));
    }
    __builtin_amdgcn_sched_barrier(0);   // keep all 8 ds_loads ahead of the WMMAs
#pragma unroll
    for (int sub = 0; sub < 4; ++sub)
      acc[sub] = wmma_f16(a, bf[sub], acc[sub]);
  };

  int s = 0;
  for (; s + 2 < S; s += 2) {            // branchless steady state: always stage 2 ahead
    const int k0 = s << 5;
    const uint32_t curB = (s & 2) ? (uint32_t)(2 * BUFB) : 0u;
    const uint32_t nxtB = curB ^ (uint32_t)(2 * BUFB);

    // ---- k-step s ----
    STAGE(lds_base + nxtB, k0 + 64)
    asm volatile("s_wait_asynccnt 0x2" ::: "memory");
    BARRIER();
    kstep(ldsf + (curB >> 1), cat16(a0lo, a0hi));
    a0lo = *(const v8h*)(arow + k0 + 64);
    a0hi = *(const v8h*)(arow + k0 + 80);
    __builtin_prefetch(arow + k0 + 512, 0, 0);

    // ---- k-step s+1 ----
    STAGE(lds_base + nxtB + BUFB, k0 + 96)
    asm volatile("s_wait_asynccnt 0x2" ::: "memory");
    BARRIER();
    kstep(ldsf + (curB >> 1) + BUFH, cat16(a1lo, a1hi));
    a1lo = *(const v8h*)(arow + k0 + 96);
    a1hi = *(const v8h*)(arow + k0 + 112);
  }
  {                                      // peeled tail: last two k-steps, nothing to stage
    const uint32_t curB = (s & 2) ? (uint32_t)(2 * BUFB) : 0u;
    asm volatile("s_wait_asynccnt 0x1" ::: "memory");
    BARRIER();
    kstep(ldsf + (curB >> 1), cat16(a0lo, a0hi));
    asm volatile("s_wait_asynccnt 0x0" ::: "memory");
    BARRIER();
    kstep(ldsf + (curB >> 1) + BUFH, cat16(a1lo, a1hi));
  }
#undef STAGE
#undef BARRIER

  const int mb = tm * 16 + 8 * hs;
#pragma unroll
  for (int sub = 0; sub < 4; ++sub) {
    int n = tq * 64 + sub * 16 + hl;
    float bn = bias[n];
#pragma unroll
    for (int r = 0; r < 8; ++r) {
      float v = acc[sub][r] + bn;
      v = v > 0.0f ? v : 0.0f;
      C[(size_t)(mb + r) * ldc + n] = (_Float16)v;
    }
  }
}

// ---------------- gather + pairwise interaction ---------------------------------------------
// 4 waves/block, 1 sample/wave. Features F(32x128) f16 in LDS (rows 27..31 zero).
// Z = F @ F^T via WMMA (B operand = F^T => column n is F row n, contiguous in LDS).
__global__ __launch_bounds__(128)
void Dlrm_interact(const _Float16* __restrict__ bot, const int* __restrict__ sidx,
                   const float* __restrict__ emb, _Float16* __restrict__ topin) {
  __shared__ _Float16 F[4][32][136];   // +8 halves pad: rows 16B-aligned, bank-staggered
  int wv = threadIdx.x >> 5;
  int lane = threadIdx.x & 31;
  int b = blockIdx.x * 4 + wv;

  // row 0: bottom MLP output (already f16)
  *(v4h*)&F[wv][0][lane * 4] = *(const v4h*)(bot + (size_t)b * 128 + lane * 4);
  // rows 1..26: embedding gather, f32 -> f16
  for (int t = 0; t < NTAB; ++t) {
    int idx = sidx[t * BATCH + b];
    const float4 f = ((const float4*)(emb + ((size_t)t * VOCAB + (size_t)idx) * EDIM))[lane];
    v4h h = { (_Float16)f.x, (_Float16)f.y, (_Float16)f.z, (_Float16)f.w };
    *(v4h*)&F[wv][1 + t][lane * 4] = h;
  }
  // rows 27..31: zero pad
  v4h z4 = {};
#pragma unroll
  for (int t = 27; t < 32; ++t) *(v4h*)&F[wv][t][lane * 4] = z4;
  __syncthreads();

  int hl = lane & 15;
  int hs = lane >> 4;
  const int MT[3] = {0, 1, 1};
  const int NT[3] = {0, 0, 1};
  _Float16* trow = topin + (size_t)b * 512;

#pragma unroll
  for (int tt = 0; tt < 3; ++tt) {
    const _Float16* ar = &F[wv][MT[tt] * 16 + hl][hs * 8];
    const _Float16* br = &F[wv][NT[tt] * 16 + hl][hs * 8];
    v8f acc = {};
#pragma unroll
    for (int k0 = 0; k0 < 128; k0 += 32) {
      v16h a = cat16(*(const v8h*)(ar + k0), *(const v8h*)(ar + k0 + 16));
      v16h bm = cat16(*(const v8h*)(br + k0), *(const v8h*)(br + k0 + 16));
      acc = wmma_f16(a, bm, acc);
    }
    int j  = NT[tt] * 16 + hl;
    int ib = MT[tt] * 16 + 8 * hs;
#pragma unroll
    for (int r = 0; r < 8; ++r) {
      int i = ib + r;
      if (i <= 26 && j < i) {
        int p = (i * (i - 1)) / 2 + j;           // strictly-lower-tri packing
        trow[128 + p] = (_Float16)acc[r];
      }
    }
  }

  // top_in = [bot_out(128) | interact(351) | zeros(33)]
  ((v4h*)trow)[lane] = *(const v4h*)&F[wv][0][lane * 4];
  if (lane < 33) trow[479 + lane] = (_Float16)0.0f;
}

// ---------------- final layer: (B,256) . w(256) + bias, relu, fp32 out ----------------------
__global__ void Dlrm_toplast(const _Float16* __restrict__ h, const _Float16* __restrict__ w,
                             const float* __restrict__ bias, float* __restrict__ out) {
  int wv = threadIdx.x >> 5;
  int lane = threadIdx.x & 31;
  int b = blockIdx.x * 8 + wv;
  v8h hv = *(const v8h*)(h + (size_t)b * 256 + lane * 8);
  v8h wvv = *(const v8h*)(w + lane * 8);
  float s = 0.0f;
#pragma unroll
  for (int j = 0; j < 8; ++j) s += (float)hv[j] * (float)wvv[j];
#pragma unroll
  for (int off = 16; off > 0; off >>= 1) s += __shfl_xor(s, off, 32);
  if (lane == 0) {
    float v = s + bias[0];
    out[b] = v > 0.0f ? v : 0.0f;
  }
}

// ---------------- workspace layout (bytes, 256-aligned) -------------------------------------
#define WT_BOT2 0u
#define WT_BOT3 262144u
#define WT_TOP1 327680u
#define WT_TOP2 1376256u
#define WT_TOP3 3473408u
#define WT_TOP4 4521984u
#define WT_TOP5 4784128u
#define OFF_H1  4784640u          // B*512 f16
#define OFF_H2  21561856u         // B*256 f16
#define OFF_BOT 29950464u         // B*128 f16
#define OFF_TI  34144768u         // B*512 f16 (padded top input)
#define OFF_T1  50921984u         // B*1024 f16
#define OFF_T2  OFF_H1            // alias: h1/h2/bot/topin dead when t2 is computed
#define OFF_T3  OFF_T1            // alias: t1 dead when t3 is computed
#define OFF_T4  (OFF_T1 + 16777216u)

static void launch_gemm(const _Float16* A, int lda, const _Float16* Wt, int ldb,
                        const float* bias, _Float16* C, int ldc,
                        int M, int N, int K, hipStream_t s) {
  dim3 grid(N / 64, M / 128);
  Dlrm_gemm<<<grid, dim3(256), 0, s>>>(A, lda, Wt, ldb, bias, C, ldc, K);
}

static void launch_wconv(const float* W, _Float16* Wt, int K, int N, int Kpad, hipStream_t s) {
  size_t total = (size_t)N * Kpad;
  Dlrm_wconv<<<dim3((unsigned)((total + 255) / 256)), dim3(256), 0, s>>>(W, Wt, K, N, Kpad);
}

extern "C" void kernel_launch(void* const* d_in, const int* in_sizes, int n_in,
                              void* d_out, int out_size, void* d_ws, size_t ws_size,
                              hipStream_t stream) {
  const float* dense = (const float*)d_in[0];
  const int*   sidx  = (const int*)d_in[1];
  /* d_in[2] sparse_offs: identity (bag size 1) -> unused */
  const float* emb   = (const float*)d_in[3];
  const float* bW0 = (const float*)d_in[4];   // 13x512
  const float* bW1 = (const float*)d_in[5];   // 512x256
  const float* bW2 = (const float*)d_in[6];   // 256x128
  const float* bb0 = (const float*)d_in[7];
  const float* bb1 = (const float*)d_in[8];
  const float* bb2 = (const float*)d_in[9];
  const float* tW0 = (const float*)d_in[10];  // 479x1024
  const float* tW1 = (const float*)d_in[11];  // 1024x1024
  const float* tW2 = (const float*)d_in[12];  // 1024x512
  const float* tW3 = (const float*)d_in[13];  // 512x256
  const float* tW4 = (const float*)d_in[14];  // 256x1
  const float* tb0 = (const float*)d_in[15];
  const float* tb1 = (const float*)d_in[16];
  const float* tb2 = (const float*)d_in[17];
  const float* tb3 = (const float*)d_in[18];
  const float* tb4 = (const float*)d_in[19];
  (void)in_sizes; (void)n_in; (void)out_size; (void)ws_size;

  char* ws = (char*)d_ws;
  _Float16* wtB2 = (_Float16*)(ws + WT_BOT2);
  _Float16* wtB3 = (_Float16*)(ws + WT_BOT3);
  _Float16* wtT0 = (_Float16*)(ws + WT_TOP1);
  _Float16* wtT1 = (_Float16*)(ws + WT_TOP2);
  _Float16* wtT2 = (_Float16*)(ws + WT_TOP3);
  _Float16* wtT3 = (_Float16*)(ws + WT_TOP4);
  _Float16* wtT4 = (_Float16*)(ws + WT_TOP5);
  _Float16* h1   = (_Float16*)(ws + OFF_H1);
  _Float16* h2   = (_Float16*)(ws + OFF_H2);
  _Float16* bot  = (_Float16*)(ws + OFF_BOT);
  _Float16* ti   = (_Float16*)(ws + OFF_TI);
  _Float16* t1   = (_Float16*)(ws + OFF_T1);
  _Float16* t2   = (_Float16*)(ws + OFF_T2);
  _Float16* t3   = (_Float16*)(ws + OFF_T3);
  _Float16* t4   = (_Float16*)(ws + OFF_T4);

  // weight convert + transpose (K padded to mult of 32; top W0 padded 479->512)
  launch_wconv(bW1, wtB2, 512, 256, 512, stream);
  launch_wconv(bW2, wtB3, 256, 128, 256, stream);
  launch_wconv(tW0, wtT0, 479, 1024, 512, stream);
  launch_wconv(tW1, wtT1, 1024, 1024, 1024, stream);
  launch_wconv(tW2, wtT2, 1024, 512, 1024, stream);
  launch_wconv(tW3, wtT3, 512, 256, 512, stream);
  launch_wconv(tW4, wtT4, 256, 1, 256, stream);

  // bottom MLP
  Dlrm_bot1<<<dim3(BATCH * 512 / 256), dim3(256), 0, stream>>>(dense, bW0, bb0, h1);
  launch_gemm(h1, 512, wtB2, 512, bb1, h2, 256, BATCH, 256, 512, stream);
  launch_gemm(h2, 256, wtB3, 256, bb2, bot, 128, BATCH, 128, 256, stream);

  // embedding gather + pairwise interaction + top-input assembly
  Dlrm_interact<<<dim3(BATCH / 4), dim3(128), 0, stream>>>(bot, sidx, emb, ti);

  // top MLP
  launch_gemm(ti, 512, wtT0, 512, tb0, t1, 1024, BATCH, 1024, 512, stream);
  launch_gemm(t1, 1024, wtT1, 1024, tb1, t2, 1024, BATCH, 1024, 1024, stream);
  launch_gemm(t2, 1024, wtT2, 1024, tb2, t3, 512, BATCH, 512, 1024, stream);
  launch_gemm(t3, 512, wtT3, 512, tb3, t4, 256, BATCH, 256, 512, stream);
  Dlrm_toplast<<<dim3(BATCH / 8), dim3(256), 0, stream>>>(t4, wtT4, tb4, (float*)d_out);
}
